// MultiHeadTemporalAttention_55473797595853
// MI455X (gfx1250) — compile-verified, matrix operand order checked by reading, not run on hardware
//
#include <hip/hip_runtime.h>
#include <hip/hip_bf16.h>

// ---------------------------------------------------------------------------
// MultiHeadTemporalAttention for gfx1250 (MI455X), bf16 WMMA pipeline with
// async global->LDS staging (ASYNCcnt) in the projection GEMMs.
// ---------------------------------------------------------------------------

#define BATCH 64
#define SEQ   200
#define DM    1024
#define NH    16
#define DKH   64
#define SPAD  224           // SEQ padded to multiple of 32 for WMMA K loops
#define QTILES 13           // ceil(200/16)

typedef __attribute__((ext_vector_type(16))) __bf16 v16bf;
typedef __attribute__((ext_vector_type(8)))  float  v8f;

struct alignas(16) U128 { unsigned int x, y, z, w; };
struct alignas(16) F128 { float a, b, c, d; };

union Frag {
    v16bf v;
    U128  q[2];
    unsigned short u[16];
};

__device__ __forceinline__ unsigned short f32_to_bf16(float f) {
    unsigned int u = __float_as_uint(f);
    unsigned int r = u + 0x7FFFu + ((u >> 16) & 1u);   // round-to-nearest-even
    return (unsigned short)(r >> 16);
}

// Async 16-byte global -> LDS copy (GLOBAL_LOAD_ASYNC_TO_LDS_B128, ASYNCcnt).
__device__ __forceinline__ void async_copy_b128(unsigned lds_off, const void* gptr) {
    asm volatile("global_load_async_to_lds_b128 %0, %1, off"
                 :: "v"(lds_off), "v"(gptr) : "memory");
}
__device__ __forceinline__ void wait_async0() {
    asm volatile("s_wait_asynccnt 0" ::: "memory");
}

// ---------------------------------------------------------------------------
// fp32 -> bf16 elementwise convert
// ---------------------------------------------------------------------------
__global__ void cvt_f32_bf16_kernel(const float* __restrict__ in,
                                    unsigned short* __restrict__ out,
                                    long long n) {
    long long i = (long long)blockIdx.x * blockDim.x + threadIdx.x;
    long long stride = (long long)gridDim.x * blockDim.x;
    for (; i < n; i += stride) out[i] = f32_to_bf16(in[i]);
}

// ---------------------------------------------------------------------------
// GEMM: Y[m,n] = sum_k X[m,k] * W[n,k] + bias[n]
// X: [12800,1024] bf16 row-major; W: [1024,1024] bf16 row-major.
// Block = 256 threads (8 waves) -> 256(M) x 64(N).  Wave -> 32(M) x 64(N):
// 8 accumulators, 8 WMMAs per K-step of 32.
// B tile (64x32 bf16 = 4KB) is staged in LDS, double-buffered, via
// global_load_async_to_lds_b128; all 8 waves consume it from LDS.
// grid = (1024/64, 12800/256).
// ---------------------------------------------------------------------------
template <bool BF16OUT>
__global__ __launch_bounds__(256)
void gemm_bias_kernel(const unsigned short* __restrict__ X,
                      const unsigned short* __restrict__ W,
                      const float* __restrict__ bias,
                      unsigned short* __restrict__ outb,
                      float* __restrict__ outf) {
    const int lane = threadIdx.x & 31;
    const int wave = threadIdx.x >> 5;
    const int l16  = lane & 15;
    const int half = (lane >> 4) & 1;
    const int m0 = blockIdx.y * 256 + wave * 32;
    const int n0 = blockIdx.x * 64;

    __shared__ __align__(16) unsigned short bstage[2][64 * 32];

    // --- async B staging setup: thread t copies row t/4, 16B chunk t%4 ---
    const int trow = threadIdx.x >> 2;     // 0..63  (B row = W row n0+trow)
    const int tchk = threadIdx.x & 3;      // 0..3   (16B chunk within 32 elems)
    const unsigned short* bsrc = W + (size_t)(n0 + trow) * DM + tchk * 8;
    unsigned lds_dst[2];
    lds_dst[0] = (unsigned)(size_t)&bstage[0][trow * 32 + tchk * 8];
    lds_dst[1] = (unsigned)(size_t)&bstage[1][trow * 32 + tchk * 8];

    v8f acc[2][4];
#pragma unroll
    for (int t = 0; t < 4; ++t) {
        float bval = bias[n0 + t * 16 + l16];
#pragma unroll
        for (int mh = 0; mh < 2; ++mh)
#pragma unroll
            for (int i = 0; i < 8; ++i) acc[mh][t][i] = bval;
    }

    // A fragment bases: lane holds rows m0+l16 and m0+16+l16;
    // K chunks {half*8 .. +7} and {+16 .. +23}.
    const unsigned short* aptr0 = X + (size_t)(m0 + l16) * DM + half * 8;
    const unsigned short* aptr1 = X + (size_t)(m0 + 16 + l16) * DM + half * 8;

    // prime buffer 0 with k0 = 0
    async_copy_b128(lds_dst[0], bsrc);
    wait_async0();
    __syncthreads();

    for (int kt = 0; kt < DM / 32; ++kt) {
        const int k0  = kt * 32;
        const int cur = kt & 1;

        // kick off next K-step's B tile into the other buffer
        if (kt + 1 < DM / 32)
            async_copy_b128(lds_dst[cur ^ 1], bsrc + (k0 + 32));
        if (kt + 2 < DM / 32)
            __builtin_prefetch(aptr0 + k0 + 64, 0, 1);   // global_prefetch_b8

        Frag a0, a1;
        a0.q[0] = *(const U128*)(aptr0 + k0);
        a0.q[1] = *(const U128*)(aptr0 + k0 + 16);
        a1.q[0] = *(const U128*)(aptr1 + k0);
        a1.q[1] = *(const U128*)(aptr1 + k0 + 16);

#pragma unroll
        for (int t = 0; t < 4; ++t) {
            Frag bfr;   // B col = n0 + t*16 + l16  ->  LDS row t*16+l16
            const unsigned short* bp = &bstage[cur][(t * 16 + l16) * 32 + half * 8];
            bfr.q[0] = *(const U128*)bp;
            bfr.q[1] = *(const U128*)(bp + 16);
            acc[0][t] = __builtin_amdgcn_wmma_f32_16x16x32_bf16(
                false, a0.v, false, bfr.v, (short)0, acc[0][t], false, false);
            acc[1][t] = __builtin_amdgcn_wmma_f32_16x16x32_bf16(
                false, a1.v, false, bfr.v, (short)0, acc[1][t], false, false);
        }

        wait_async0();      // next buffer resident before anyone flips to it
        __syncthreads();    // all waves done reading current buffer
    }

#pragma unroll
    for (int mh = 0; mh < 2; ++mh) {
#pragma unroll
        for (int t = 0; t < 4; ++t) {
            const int col = n0 + t * 16 + l16;
#pragma unroll
            for (int r = 0; r < 8; ++r) {
                const int row = m0 + mh * 16 + r + half * 8;
                const size_t o = (size_t)row * DM + col;
                if (BF16OUT) outb[o] = f32_to_bf16(acc[mh][t][r]);
                else         outf[o] = acc[mh][t][r];
            }
        }
    }
}

// ---------------------------------------------------------------------------
// Transpose + pad V: [b,s,h,d] bf16 -> Vt [b,h,d, SPAD] bf16 (zero pad s>=200)
// ---------------------------------------------------------------------------
__global__ void build_vt_kernel(const unsigned short* __restrict__ V,
                                unsigned short* __restrict__ Vt) {
    long long idx = (long long)blockIdx.x * blockDim.x + threadIdx.x;
    const long long total = (long long)BATCH * NH * DKH * SPAD;
    if (idx >= total) return;
    int k = (int)(idx % SPAD);
    long long rest = idx / SPAD;
    int d = (int)(rest % DKH); rest /= DKH;
    int h = (int)(rest % NH);
    int b = (int)(rest / NH);
    unsigned short v = 0;
    if (k < SEQ) v = V[((size_t)(b * SEQ + k)) * DM + h * DKH + d];
    Vt[idx] = v;
}

// ---------------------------------------------------------------------------
// Attention: one wave per (b, h, 16-query strip).
//   scores = Q K^T / 8 + bias, mask, softmax  -> attn (to d_out + LDS)
//   ctx    = attn @ V                          -> ctx buffer (bf16)
// grid = (13, NH, BATCH), block = 32.
// ---------------------------------------------------------------------------
__global__ __launch_bounds__(32)
void attention_kernel(const unsigned short* __restrict__ Q,
                      const unsigned short* __restrict__ K,
                      const unsigned short* __restrict__ Vt,
                      const int* __restrict__ mask,
                      const float* __restrict__ tbias,
                      float* __restrict__ attn_out,
                      unsigned short* __restrict__ ctx) {
    const int b  = blockIdx.z;
    const int h  = blockIdx.y;
    const int q0 = blockIdx.x * 16;
    const int lane = threadIdx.x & 31;
    const int l16  = lane & 15;
    const int half = (lane >> 4) & 1;

    __shared__ __align__(16) float smem[16][SPAD];

    // ---- scores = Q K^T ----
    v8f acc[QTILES];
#pragma unroll
    for (int j = 0; j < QTILES; ++j)
#pragma unroll
        for (int i = 0; i < 8; ++i) acc[j][i] = 0.0f;

    int qrow = q0 + l16; if (qrow > SEQ - 1) qrow = SEQ - 1;
    Frag qf[2];
#pragma unroll
    for (int kk = 0; kk < 2; ++kk) {
        const unsigned short* p =
            Q + (size_t)(b * SEQ + qrow) * DM + h * DKH + kk * 32 + half * 8;
        qf[kk].q[0] = *(const U128*)p;
        qf[kk].q[1] = *(const U128*)(p + 16);
    }

#pragma unroll
    for (int j = 0; j < QTILES; ++j) {
        int key = j * 16 + l16; if (key > SEQ - 1) key = SEQ - 1;
#pragma unroll
        for (int kk = 0; kk < 2; ++kk) {
            Frag bfr;
            const unsigned short* p =
                K + (size_t)(b * SEQ + key) * DM + h * DKH + kk * 32 + half * 8;
            bfr.q[0] = *(const U128*)p;
            bfr.q[1] = *(const U128*)(p + 16);
            acc[j] = __builtin_amdgcn_wmma_f32_16x16x32_bf16(
                false, qf[kk].v, false, bfr.v, (short)0, acc[j], false, false);
        }
    }

    // ---- bias + mask + softmax (D layout: row r+half*8, col l16 per tile) ----
#pragma unroll
    for (int r = 0; r < 8; ++r) {
        const int q = q0 + r + half * 8;
        float mx = -3.0e38f;
#pragma unroll
        for (int j = 0; j < QTILES; ++j) {
            const int col = j * 16 + l16;
            float s = acc[j][r] * 0.125f;               // / sqrt(64)
            if (q < SEQ && col < SEQ) {
                s += tbias[h * (SEQ * SEQ) + q * SEQ + col];
                if (mask[(size_t)b * SEQ * SEQ + q * SEQ + col] == 0) s = -1e9f;
            } else {
                s = -1e9f;
            }
            acc[j][r] = s;
            mx = fmaxf(mx, s);
        }
        for (int off = 1; off < 16; off <<= 1)
            mx = fmaxf(mx, __shfl_xor(mx, off, 32));
        float sum = 0.0f;
#pragma unroll
        for (int j = 0; j < QTILES; ++j) {
            float p = __expf(acc[j][r] - mx);
            acc[j][r] = p;
            sum += p;
        }
        for (int off = 1; off < 16; off <<= 1)
            sum += __shfl_xor(sum, off, 32);
        const float inv = 1.0f / sum;
#pragma unroll
        for (int j = 0; j < QTILES; ++j) acc[j][r] *= inv;
    }

    // ---- write attn to d_out and to LDS (pad cols 208..223 with zero) ----
    for (int i = lane; i < 256; i += 32) smem[i >> 4][208 + (i & 15)] = 0.0f;
#pragma unroll
    for (int j = 0; j < QTILES; ++j) {
#pragma unroll
        for (int r = 0; r < 8; ++r) {
            const int ql  = r + half * 8;
            const int col = j * 16 + l16;
            const float p = acc[j][r];
            smem[ql][col] = p;
            const int q = q0 + ql;
            if (q < SEQ && col < SEQ)
                attn_out[(((size_t)b * NH + h) * SEQ + q) * SEQ + col] = p;
        }
    }
    __syncthreads();

    // ---- ctx = attn @ V  (A from LDS transposed view, B from Vt) ----
#pragma unroll
    for (int t = 0; t < 4; ++t) {
        v8f c;
#pragma unroll
        for (int i = 0; i < 8; ++i) c[i] = 0.0f;
        for (int ks = 0; ks < SPAD / 32; ++ks) {
            const int koff = ks * 32 + half * 8;
            const float* rowp = &smem[l16][0];
            F128 f0 = *(const F128*)(rowp + koff);
            F128 f1 = *(const F128*)(rowp + koff + 4);
            F128 f2 = *(const F128*)(rowp + koff + 16);
            F128 f3 = *(const F128*)(rowp + koff + 20);
            Frag af;
            af.u[0]  = f32_to_bf16(f0.a); af.u[1]  = f32_to_bf16(f0.b);
            af.u[2]  = f32_to_bf16(f0.c); af.u[3]  = f32_to_bf16(f0.d);
            af.u[4]  = f32_to_bf16(f1.a); af.u[5]  = f32_to_bf16(f1.b);
            af.u[6]  = f32_to_bf16(f1.c); af.u[7]  = f32_to_bf16(f1.d);
            af.u[8]  = f32_to_bf16(f2.a); af.u[9]  = f32_to_bf16(f2.b);
            af.u[10] = f32_to_bf16(f2.c); af.u[11] = f32_to_bf16(f2.d);
            af.u[12] = f32_to_bf16(f3.a); af.u[13] = f32_to_bf16(f3.b);
            af.u[14] = f32_to_bf16(f3.c); af.u[15] = f32_to_bf16(f3.d);
            Frag bfr;
            const unsigned short* p =
                Vt + (((size_t)b * NH + h) * DKH + t * 16 + l16) * SPAD + koff;
            bfr.q[0] = *(const U128*)p;
            bfr.q[1] = *(const U128*)(p + 16);
            c = __builtin_amdgcn_wmma_f32_16x16x32_bf16(
                false, af.v, false, bfr.v, (short)0, c, false, false);
        }
#pragma unroll
        for (int r = 0; r < 8; ++r) {
            const int q = q0 + r + half * 8;
            if (q < SEQ) {
                const int d = t * 16 + l16;
                ctx[((size_t)(b * SEQ + q)) * DM + h * DKH + d] = f32_to_bf16(c[r]);
            }
        }
    }
}

// ---------------------------------------------------------------------------
// Launch
// ---------------------------------------------------------------------------
extern "C" void kernel_launch(void* const* d_in, const int* in_sizes, int n_in,
                              void* d_out, int out_size, void* d_ws, size_t ws_size,
                              hipStream_t stream) {
    const float* x    = (const float*)d_in[0];
    const int*   mask = (const int*)d_in[1];
    const float* wq   = (const float*)d_in[2];
    const float* bq   = (const float*)d_in[3];
    const float* wk   = (const float*)d_in[4];
    const float* bk   = (const float*)d_in[5];
    const float* wv   = (const float*)d_in[6];
    const float* bv   = (const float*)d_in[7];
    const float* wo   = (const float*)d_in[8];
    const float* bo   = (const float*)d_in[9];
    const float* tb   = (const float*)d_in[10];

    const long long XN = (long long)BATCH * SEQ * DM;   // 13,107,200
    const long long WN = (long long)DM * DM;            // 1,048,576
    const long long VTN = (long long)BATCH * NH * DKH * SPAD;

    char* w = (char*)d_ws;
    auto alloc = [&](size_t bytes) -> char* {
        char* p = w;
        w += (bytes + 255) & ~(size_t)255;
        return p;
    };
    unsigned short* Xb  = (unsigned short*)alloc((size_t)XN * 2);
    unsigned short* Wqb = (unsigned short*)alloc((size_t)WN * 2);
    unsigned short* Wkb = (unsigned short*)alloc((size_t)WN * 2);
    unsigned short* Wvb = (unsigned short*)alloc((size_t)WN * 2);
    unsigned short* Wob = (unsigned short*)alloc((size_t)WN * 2);
    unsigned short* Qb  = (unsigned short*)alloc((size_t)XN * 2);
    unsigned short* Kb  = (unsigned short*)alloc((size_t)XN * 2);
    unsigned short* Vb  = (unsigned short*)alloc((size_t)XN * 2);
    unsigned short* Vt  = (unsigned short*)alloc((size_t)VTN * 2);
    unsigned short* Cb  = (unsigned short*)alloc((size_t)XN * 2);

    float* out_f  = (float*)d_out;
    float* attn_f = out_f + XN;

    // 1) fp32 -> bf16 converts
    cvt_f32_bf16_kernel<<<(int)((XN + 255) / 256), 256, 0, stream>>>(x, Xb, XN);
    cvt_f32_bf16_kernel<<<(int)((WN + 255) / 256), 256, 0, stream>>>(wq, Wqb, WN);
    cvt_f32_bf16_kernel<<<(int)((WN + 255) / 256), 256, 0, stream>>>(wk, Wkb, WN);
    cvt_f32_bf16_kernel<<<(int)((WN + 255) / 256), 256, 0, stream>>>(wv, Wvb, WN);
    cvt_f32_bf16_kernel<<<(int)((WN + 255) / 256), 256, 0, stream>>>(wo, Wob, WN);

    // 2) QKV projections (WMMA + async LDS staging)
    dim3 ggrid(DM / 64, (BATCH * SEQ) / 256);
    gemm_bias_kernel<true><<<ggrid, 256, 0, stream>>>(Xb, Wqb, bq, Qb, nullptr);
    gemm_bias_kernel<true><<<ggrid, 256, 0, stream>>>(Xb, Wkb, bk, Kb, nullptr);
    gemm_bias_kernel<true><<<ggrid, 256, 0, stream>>>(Xb, Wvb, bv, Vb, nullptr);

    // 3) V transpose/pad
    build_vt_kernel<<<(int)((VTN + 255) / 256), 256, 0, stream>>>(Vb, Vt);

    // 4) attention (WMMA scores + softmax + WMMA ctx)
    attention_kernel<<<dim3(QTILES, NH, BATCH), 32, 0, stream>>>(
        Qb, Kb, Vt, mask, tb, attn_f, Cb);

    // 5) output projection (WMMA, fp32 out straight into d_out)
    gemm_bias_kernel<false><<<ggrid, 256, 0, stream>>>(Cb, Wob, bo, nullptr, out_f);
}
